// GumbelNeRFBig_53403623358652
// MI455X (gfx1250) — matrix operand order repaired and checked by reading, not compiled, not run on hardware
//
#include <hip/hip_runtime.h>
#include <hip/hip_bf16.h>

// ---------------------------------------------------------------------------
// Gumbel-NeRF MoE, fused for CDNA5 (gfx1250, wave32).
//  - All GEMMs on v_wmma_f32_16x16x32_bf16 (fp32 accumulate).
//  - Weight panels double-buffered in LDS, staged with
//    GLOBAL_LOAD_ASYNC_TO_LDS_B128 (ASYNCcnt) overlapped with WMMA.
//  - B fragments double-buffered in VGPRs to hide DS latency behind the XDL.
//  - Whole per-tile network (128 points) lives in LDS: HBM traffic ~11 MB.
// ---------------------------------------------------------------------------

#define NPTS   131072
#define NEXP   4
#define NLYR   4
#define HDIM   256
#define RGBH   128
#define MT     128          // points per workgroup
#define ASTR   264          // activation LDS row stride (ushort), padded
#define WSTR   40           // staged-weight LDS row stride (ushort), padded
#define VSTR   40           // viewdir-PE LDS row stride (ushort)
#define KPAD_ENC 64         // 63 -> 64
#define KPAD_RGB 288        // 283 -> 288
#define PANEL  (256 * WSTR) // one staged weight panel (ushorts)

// d_ws layout (ushort/bf16 elements)
#define WS_ENC  0                                   // [256 n][64 k]
#define WS_MLP  (WS_ENC + 256 * 64)                 // [E*L][256 n][256 k]
#define WS_RGB1 (WS_MLP + NEXP * NLYR * HDIM * HDIM)// [E][128 n][288 k]

typedef __attribute__((ext_vector_type(16))) __bf16 v16bf;
typedef __attribute__((ext_vector_type(8)))  float  v8f;

union Frag { uint4 q[2]; v16bf v; };

__device__ __forceinline__ unsigned short f2bf(float f) {
  unsigned u = __float_as_uint(f);
  u += 0x7FFFu + ((u >> 16) & 1u);      // round-to-nearest-even
  return (unsigned short)(u >> 16);
}
__device__ __forceinline__ float bf2f(unsigned short s) {
  return __uint_as_float(((unsigned)s) << 16);
}

// ---------------------------------------------------------------------------
// Weight pre-transpose/convert kernels: fp32 [K][N] -> bf16 [N][Kpad]
// ---------------------------------------------------------------------------
__global__ void conv_enc(const float* __restrict__ w, unsigned short* __restrict__ o) {
  int i = blockIdx.x * 256 + threadIdx.x;           // 256*64
  if (i >= 256 * 64) return;
  int n = i >> 6, k = i & 63;
  float v = (k < 63) ? w[k * 256 + n] : 0.f;
  o[WS_ENC + n * 64 + k] = f2bf(v);
}
__global__ void conv_mlp(const float* __restrict__ w, unsigned short* __restrict__ o) {
  int i = blockIdx.x * 256 + threadIdx.x;           // 16*256*256 = 1048576
  if (i >= NEXP * NLYR * HDIM * HDIM) return;
  int el = i >> 16; int r = i & 65535; int n = r >> 8; int k = r & 255;
  o[WS_MLP + el * 65536 + n * 256 + k] = f2bf(w[el * 65536 + k * 256 + n]);
}
__global__ void conv_rgb1(const float* __restrict__ w, unsigned short* __restrict__ o) {
  int i = blockIdx.x * 256 + threadIdx.x;           // 4*128*288 = 147456
  if (i >= NEXP * RGBH * KPAD_RGB) return;
  int e = i / (RGBH * KPAD_RGB);
  int r = i % (RGBH * KPAD_RGB);
  int n = r / KPAD_RGB, k = r % KPAD_RGB;
  float v = (k < 283) ? w[(e * 283 + k) * RGBH + n] : 0.f;
  o[WS_RGB1 + (e * RGBH + n) * KPAD_RGB + k] = f2bf(v);
}

// ---------------------------------------------------------------------------
// Async LDS staging of one [32 k][nout n] panel (pre-transposed [n][kpad]).
// GVS form: lds[vdst + off] = mem[saddr + vaddr + off]; tracked by ASYNCcnt.
// ---------------------------------------------------------------------------
__device__ __forceinline__ void stage_async(const unsigned short* __restrict__ wT,
                                            int kpad, int k0, int nout,
                                            unsigned short* dst) {
  const int t = threadIdx.x;
  if (t < nout) {
    unsigned lds  = (unsigned)(unsigned long long)(dst + t * WSTR); // LDS byte addr
    unsigned gofs = (unsigned)((t * kpad + k0) * 2);                // global byte off
    asm volatile(
        "global_load_async_to_lds_b128 %0, %1, %2\n\t"
        "global_load_async_to_lds_b128 %0, %1, %2 offset:16\n\t"
        "global_load_async_to_lds_b128 %0, %1, %2 offset:32\n\t"
        "global_load_async_to_lds_b128 %0, %1, %2 offset:48"
        :
        : "v"(lds), "v"(gofs), "s"(wT)
        : "memory");
  }
}
__device__ __forceinline__ void wait_async0() {
  asm volatile("s_wait_asynccnt 0" ::: "memory");
}

// ---------------------------------------------------------------------------
// Fragment loaders (ISA 7.12.2 VGPR layouts, wave32)
// ---------------------------------------------------------------------------
// A 16x32 bf16: lanes 0-15 hold K 0..7 & 16..23; lanes 16-31 hold K 8..15 & 24..31
__device__ __forceinline__ v16bf load_a(const unsigned short* act, int stride,
                                        int row, int k0, int hl) {
  Frag f;
  const uint4* p = (const uint4*)(act + row * stride + k0 + 8 * hl);
  f.q[0] = p[0];          // K: k0+8h .. +7
  f.q[1] = p[2];          // K: k0+8h+16 .. +23
  return f.v;
}
// B 32x16 bf16: lanes 0-15: col=lane, K 0..7 & 16..23; lanes 16-31: col=lane-16, K 8..15 & 24..31
__device__ __forceinline__ v16bf load_b(const unsigned short* wbuf, int ncol, int hl) {
  Frag f;
  const uint4* p = (const uint4*)(wbuf + ncol * WSTR + 8 * hl);
  f.q[0] = p[0];
  f.q[1] = p[2];
  return f.v;
}

// ---------------------------------------------------------------------------
// One dense layer: out[128 x NT*16] = act[128 x 32*kTot] @ W + b, opt. relu.
// Double-buffered weight panels (async stage of panel ks+1 overlaps the WMMAs
// on panel ks; one barrier per K-step) and double-buffered B fragments in
// VGPRs (load B(nt+1) before WMMA(nt) so DS latency hides behind the XDL).
// K-steps >= kMain read activations from aTail (RGB concat [shape|vd_pe]).
// ---------------------------------------------------------------------------
template<int NT>
__device__ void gemm_layer(const unsigned short* aBuf, int aStride,
                           const unsigned short* aTail, int aTailStride, int kMain,
                           int kTot, unsigned short* outBuf,
                           const unsigned short* __restrict__ wT, int kpad,
                           const float* __restrict__ bias, int relu,
                           unsigned short* wbuf) {
  const int lane = threadIdx.x & 31;
  const int wid  = threadIdx.x >> 5;
  const int hl   = lane >> 4;
  const int mr   = lane & 15;
  const int nc   = lane & 15;
  const int m0   = wid * 16;

  v8f acc[NT];
  const v8f vz = {0.f, 0.f, 0.f, 0.f, 0.f, 0.f, 0.f, 0.f};
#pragma unroll
  for (int i = 0; i < NT; i++) acc[i] = vz;

  // prologue: stage panel 0 (previous layer ended with a barrier)
  stage_async(wT, kpad, 0, NT * 16, wbuf);
  wait_async0();
  __syncthreads();

  for (int ks = 0; ks < kTot; ks++) {
    unsigned short* cur = wbuf + (ks & 1) * PANEL;
    unsigned short* nxt = wbuf + ((ks + 1) & 1) * PANEL;
    if (ks + 1 < kTot)
      stage_async(wT, kpad, (ks + 1) * 32, NT * 16, nxt);   // overlap with WMMA
    v16bf a = (ks < kMain)
                ? load_a(aBuf,  aStride,     m0 + mr, ks * 32, hl)
                : load_a(aTail, aTailStride, m0 + mr, (ks - kMain) * 32, hl);
    // B-fragment software pipeline: issue load for nt+1 before WMMA on nt.
    v16bf bcur = load_b(cur, nc, hl);
#pragma unroll
    for (int nt = 0; nt < NT; nt++) {
      v16bf bnext;
      if (nt + 1 < NT) bnext = load_b(cur, (nt + 1) * 16 + nc, hl);
      acc[nt] = __builtin_amdgcn_wmma_f32_16x16x32_bf16(
          false, a, false, bcur, (short)0, acc[nt], false, false);
      bcur = bnext;
    }
    wait_async0();                     // own async stage of nxt complete
    __syncthreads();                   // panel nxt ready; cur free after next iter
  }

  // C 16x16 f32 layout: VGPR j -> row m0 + j + 8*hl, col = lane&15
#pragma unroll
  for (int nt = 0; nt < NT; nt++) {
    int col = nt * 16 + nc;
    float bv = bias[col];
#pragma unroll
    for (int j = 0; j < 8; j++) {
      float v = acc[nt][j] + bv;
      if (relu) v = fmaxf(v, 0.f);
      outBuf[(m0 + j + 8 * hl) * ASTR + col] = f2bf(v);
    }
  }
}

// ---------------------------------------------------------------------------
// Fused main kernel: PE -> encoder -> 4x(MLP trunk + sigma + RGB head) -> gate
// ---------------------------------------------------------------------------
__global__ __launch_bounds__(256, 1)
void nerf_moe_kernel(const float* __restrict__ x,
                     const float* __restrict__ gumbel,
                     const float* __restrict__ enc_b,
                     const float* __restrict__ mlp_b,
                     const float* __restrict__ sigma_w,
                     const float* __restrict__ sigma_b,
                     const float* __restrict__ rgb_b1,
                     const float* __restrict__ rgb_w2,
                     const float* __restrict__ rgb_b2,
                     const unsigned short* __restrict__ ws,
                     float* __restrict__ out) {
  extern __shared__ unsigned short sh[];
  unsigned short* actY  = sh;                 // [128][264] bf16  (y, persists)
  unsigned short* actH  = sh + 33792;         // [128][264] bf16  (ping)
  unsigned short* actB  = sh + 67584;         // [128][264] bf16  (pong)
  unsigned short* wbuf  = sh + 101376;        // 2 x [256][40] bf16 weight panels
  unsigned short* vdbuf = sh + 121856;        // [128][40]  bf16  viewdir PE
  float* sigbuf = (float*)(sh + 126976);      // [4][128] softplus(sigma)
  float* rgbbuf = sigbuf + NEXP * MT;         // [4][128][4] rgb per expert

  const int t = threadIdx.x;
  const int pbase = blockIdx.x * MT;

  // ---- positional encoding (one thread per point) -> actH (xyz), vdbuf (vd)
  if (t < MT) {
    int p = pbase + t;
    float c0 = x[p * 6 + 0], c1 = x[p * 6 + 1], c2 = x[p * 6 + 2];
    float d0 = x[p * 6 + 3], d1 = x[p * 6 + 4], d2 = x[p * 6 + 5];
    unsigned short* ar = actH + t * ASTR;
    ar[0] = f2bf(c0); ar[1] = f2bf(c1); ar[2] = f2bf(c2);
    float sc = 1.f;
    for (int i = 0; i < 10; i++) {
      float v0 = sc * c0, v1 = sc * c1, v2 = sc * c2;
      ar[3 + i * 3 + 0]  = f2bf(sinf(v0));
      ar[3 + i * 3 + 1]  = f2bf(sinf(v1));
      ar[3 + i * 3 + 2]  = f2bf(sinf(v2));
      ar[33 + i * 3 + 0] = f2bf(cosf(v0));
      ar[33 + i * 3 + 1] = f2bf(cosf(v1));
      ar[33 + i * 3 + 2] = f2bf(cosf(v2));
      sc *= 2.f;
    }
    ar[63] = 0;                                   // pad K 63 -> 64
    unsigned short* vr = vdbuf + t * VSTR;
    vr[0] = f2bf(d0); vr[1] = f2bf(d1); vr[2] = f2bf(d2);
    sc = 1.f;
    for (int i = 0; i < 4; i++) {
      float v0 = sc * d0, v1 = sc * d1, v2 = sc * d2;
      vr[3 + i * 3 + 0]  = f2bf(sinf(v0));
      vr[3 + i * 3 + 1]  = f2bf(sinf(v1));
      vr[3 + i * 3 + 2]  = f2bf(sinf(v2));
      vr[15 + i * 3 + 0] = f2bf(cosf(v0));
      vr[15 + i * 3 + 1] = f2bf(cosf(v1));
      vr[15 + i * 3 + 2] = f2bf(cosf(v2));
      sc *= 2.f;
    }
    for (int k = 27; k < 32; k++) vr[k] = 0;      // pad K 27 -> 32
  }
  __syncthreads();                                // PE visible before encoder

  // ---- encoder: y = xyz_pe @ enc_w + enc_b   (K=64, no activation)
  gemm_layer<16>(actH, ASTR, actH, ASTR, 2, 2, actY,
                 ws + WS_ENC, KPAD_ENC, enc_b, 0, wbuf);

  for (int e = 0; e < NEXP; e++) {
    const unsigned short* mw = ws + WS_MLP + e * NLYR * HDIM * HDIM;
    const float* mb = mlp_b + e * NLYR * HDIM;
    // trunk: relu applied on inputs of layers 1..3; final layer raw
    gemm_layer<16>(actY, ASTR, actY, ASTR, 8, 8, actH, mw + 0 * HDIM * HDIM, HDIM, mb + 0 * HDIM, 1, wbuf);
    gemm_layer<16>(actH, ASTR, actH, ASTR, 8, 8, actB, mw + 1 * HDIM * HDIM, HDIM, mb + 1 * HDIM, 1, wbuf);
    gemm_layer<16>(actB, ASTR, actB, ASTR, 8, 8, actH, mw + 2 * HDIM * HDIM, HDIM, mb + 2 * HDIM, 1, wbuf);
    gemm_layer<16>(actH, ASTR, actH, ASTR, 8, 8, actB, mw + 3 * HDIM * HDIM, HDIM, mb + 3 * HDIM, 0, wbuf);

    // sigma head: softplus(shape_out @ sigma_w + sigma_b), vectorized LDS reads
    __syncthreads();
    if (t < MT) {
      float s = sigma_b[0];
      const uint4* rq = (const uint4*)(actB + t * ASTR);
#pragma unroll 4
      for (int kk = 0; kk < HDIM / 8; kk++) {
        uint4 q = rq[kk];
        unsigned w0 = q.x, w1 = q.y, w2 = q.z, w3 = q.w;
        const float* swp = sigma_w + kk * 8;
        s += bf2f((unsigned short)w0)         * swp[0];
        s += bf2f((unsigned short)(w0 >> 16)) * swp[1];
        s += bf2f((unsigned short)w1)         * swp[2];
        s += bf2f((unsigned short)(w1 >> 16)) * swp[3];
        s += bf2f((unsigned short)w2)         * swp[4];
        s += bf2f((unsigned short)(w2 >> 16)) * swp[5];
        s += bf2f((unsigned short)w3)         * swp[6];
        s += bf2f((unsigned short)(w3 >> 16)) * swp[7];
      }
      sigbuf[e * MT + t] = fmaxf(s, 0.f) + log1pf(expf(-fabsf(s)));
    }

    // rgb layer1: relu([shape_out | vd_pe] @ rgb_w1 + b1) -> actH [128][128]
    gemm_layer<8>(actB, ASTR, vdbuf, VSTR, 8, 9, actH,
                  ws + WS_RGB1 + e * RGBH * KPAD_RGB, KPAD_RGB,
                  rgb_b1 + e * RGBH, 1, wbuf);

    // rgb layer2: sigmoid(h1 @ rgb_w2 + b2), N=3 -> VALU
    __syncthreads();
    if (t < MT) {
      float a0 = rgb_b2[e * 3 + 0], a1 = rgb_b2[e * 3 + 1], a2 = rgb_b2[e * 3 + 2];
      const uint4* rq = (const uint4*)(actH + t * ASTR);
#pragma unroll 2
      for (int kk = 0; kk < RGBH / 8; kk++) {
        uint4 q = rq[kk];
        unsigned pk[4] = {q.x, q.y, q.z, q.w};
#pragma unroll
        for (int j = 0; j < 4; j++) {
          float h0 = bf2f((unsigned short)pk[j]);
          float h1 = bf2f((unsigned short)(pk[j] >> 16));
          const float* w = rgb_w2 + (e * RGBH + kk * 8 + j * 2) * 3;
          a0 += h0 * w[0] + h1 * w[3];
          a1 += h0 * w[1] + h1 * w[4];
          a2 += h0 * w[2] + h1 * w[5];
        }
      }
      float* rb = rgbbuf + (e * MT + t) * 4;
      rb[0] = 1.f / (1.f + expf(-a0));
      rb[1] = 1.f / (1.f + expf(-a1));
      rb[2] = 1.f / (1.f + expf(-a2));
    }
  }

  // ---- gumbel-softmax gating + output assembly
  __syncthreads();
  if (t < MT) {
    int p = pbase + t;
    float s[NEXP], ll[NEXP], gs[NEXP], logits[NEXP], gh[NEXP];
    float m = -3.4e38f;
    for (int e = 0; e < NEXP; e++) {
      s[e]  = sigbuf[e * MT + t];
      ll[e] = logf(s[e] + 1e-10f) / 0.166667f;
      m = fmaxf(m, ll[e]);
    }
    float lse = 0.f;
    for (int e = 0; e < NEXP; e++) lse += expf(ll[e] - m);
    lse = logf(lse);
    for (int e = 0; e < NEXP; e++) { ll[e] = ll[e] - m - lse; logits[e] = expf(ll[e]); }

    float gm = -3.4e38f, g[NEXP];
    for (int e = 0; e < NEXP; e++) { g[e] = ll[e] + gumbel[p * 4 + e]; gm = fmaxf(gm, g[e]); }
    float gsum = 0.f;
    for (int e = 0; e < NEXP; e++) { gs[e] = expf(g[e] - gm); gsum += gs[e]; }
    for (int e = 0; e < NEXP; e++) gs[e] /= gsum;

    int idx = 0; float best = gs[0];
    for (int e = 1; e < NEXP; e++) if (gs[e] > best) { best = gs[e]; idx = e; }
    for (int e = 0; e < NEXP; e++) gh[e] = ((e == idx) ? 1.f : 0.f) - gs[e] + gs[e];

    float r0 = 0.f, r1 = 0.f, r2 = 0.f;
    for (int e = 0; e < NEXP; e++) {
      const float* rb = rgbbuf + (e * MT + t) * 4;
      r0 += gh[e] * rb[0]; r1 += gh[e] * rb[1]; r2 += gh[e] * rb[2];
    }

    float* o0 = out;                       // outputs [N,4]
    float* o1 = out + (size_t)NPTS * 4;    // logits
    float* o2 = out + (size_t)NPTS * 8;    // gates_hard
    float* o3 = out + (size_t)NPTS * 12;   // gates_soft
    o0[p * 4 + 0] = r0; o0[p * 4 + 1] = r1; o0[p * 4 + 2] = r2; o0[p * 4 + 3] = s[idx];
    for (int e = 0; e < NEXP; e++) {
      o1[p * 4 + e] = logits[e];
      o2[p * 4 + e] = gh[e];
      o3[p * 4 + e] = gs[e];
    }
  }
}

// ---------------------------------------------------------------------------
extern "C" void kernel_launch(void* const* d_in, const int* in_sizes, int n_in,
                              void* d_out, int out_size, void* d_ws, size_t ws_size,
                              hipStream_t stream) {
  const float* x       = (const float*)d_in[0];
  const float* gumbel  = (const float*)d_in[1];
  const float* enc_w   = (const float*)d_in[2];
  const float* enc_b   = (const float*)d_in[3];
  const float* mlp_w   = (const float*)d_in[4];
  const float* mlp_b   = (const float*)d_in[5];
  const float* sigma_w = (const float*)d_in[6];
  const float* sigma_b = (const float*)d_in[7];
  const float* rgb_w1  = (const float*)d_in[8];
  const float* rgb_b1  = (const float*)d_in[9];
  const float* rgb_w2  = (const float*)d_in[10];
  const float* rgb_b2  = (const float*)d_in[11];
  unsigned short* ws   = (unsigned short*)d_ws;
  float* out           = (float*)d_out;

  conv_enc <<<64,   256, 0, stream>>>(enc_w,  ws);
  conv_mlp <<<4096, 256, 0, stream>>>(mlp_w,  ws);
  conv_rgb1<<<576,  256, 0, stream>>>(rgb_w1, ws);

  // LDS: 3 act buffers + 2 weight panels + vd PE (ushort) + sigma/rgb (float)
  size_t smem = (size_t)126976 * 2 + (NEXP * MT + NEXP * MT * 4) * sizeof(float);
  nerf_moe_kernel<<<NPTS / MT, 256, smem, stream>>>(
      x, gumbel, enc_b, mlp_b, sigma_w, sigma_b, rgb_b1, rgb_w2, rgb_b2, ws, out);
}
